// StandardAttention_87849261072925
// MI455X (gfx1250) — matrix operand order repaired and checked by reading, not compiled, not run on hardware
//
#include <hip/hip_runtime.h>

typedef __attribute__((ext_vector_type(16))) __bf16        v16bf;
typedef __attribute__((ext_vector_type(8)))  float         v8f;
typedef __attribute__((ext_vector_type(8)))  unsigned int  v8u;
typedef __attribute__((vector_size(16)))     int           v4i_t;

#define DEV __device__ __forceinline__

constexpr int   Bc = 4, Hc = 16, Nc = 2048, Dc = 64;
constexpr int   BM = 128;           // query rows per workgroup
constexpr int   BN = 64;            // keys per iteration
constexpr int   NWAVE = 8;          // BM / 16
constexpr int   NTHREAD = NWAVE * 32;
constexpr int   NT = Nc / BN;       // 32 key tiles
constexpr float SCALE  = 0.125f;    // 1/sqrt(64)
constexpr float LOG2E  = 1.44269504088896340736f;

// ---------- CDNA5 async global->LDS copy (ASYNCcnt-tracked) ----------
#if __has_builtin(__builtin_amdgcn_global_load_async_to_lds_b128)
DEV void g2l_b128(const float* g, float* l) {
    __builtin_amdgcn_global_load_async_to_lds_b128(
        (__attribute__((address_space(1))) v4i_t*)g,
        (__attribute__((address_space(3))) v4i_t*)l, 0, 0);
}
#else
DEV void g2l_b128(const float* g, float* l) {
    const unsigned loff =
        (unsigned)(unsigned long long)(__attribute__((address_space(3))) float*)l;
    const unsigned long long ga = (unsigned long long)g;
    asm volatile("global_load_async_to_lds_b128 %0, %1, off"
                 :: "v"(loff), "v"(ga) : "memory");
}
#endif

#if __has_builtin(__builtin_amdgcn_s_wait_asynccnt)
#define WAIT_ASYNC(n) __builtin_amdgcn_s_wait_asynccnt(n)
#else
#define WAIT_ASYNC(n) asm volatile("s_wait_asynccnt " #n ::: "memory")
#endif

// ---------- cheap fp32 -> bf16 (round-half-up) ----------
DEV unsigned short f2bf(float f) {
    return (unsigned short)((__builtin_bit_cast(unsigned, f) + 0x8000u) >> 16);
}
DEV unsigned pack2(float a, float b) {   // {bf16(b), bf16(a)} in one v_perm_b32
    const unsigned ua = __builtin_bit_cast(unsigned, a) + 0x8000u;
    const unsigned ub = __builtin_bit_cast(unsigned, b) + 0x8000u;
    return __builtin_amdgcn_perm(ub, ua, 0x07060302u);
}
DEV v8f wmma_bf16(v8u a, v8u b, v8f c) {
    return __builtin_amdgcn_wmma_f32_16x16x32_bf16(
        false, __builtin_bit_cast(v16bf, a),
        false, __builtin_bit_cast(v16bf, b),
        (short)0, c, false, false);
}

// ---------- 16-lane xor reductions via DPP16 ROW_XMASK (pure VALU) ----------
template <int CTRL>
DEV float dpp_xmask(float v) {
    const int s = __builtin_bit_cast(int, v);
    const int t = __builtin_amdgcn_update_dpp(s, s, CTRL, 0xF, 0xF, true);
    return __builtin_bit_cast(float, t);
}
DEV float redmax16(float v) {
    v = fmaxf(v, dpp_xmask<0x161>(v));   // row_xmask:1
    v = fmaxf(v, dpp_xmask<0x162>(v));   // row_xmask:2
    v = fmaxf(v, dpp_xmask<0x164>(v));   // row_xmask:4
    v = fmaxf(v, dpp_xmask<0x168>(v));   // row_xmask:8
    return v;
}
DEV float redsum16(float v) {
    v += dpp_xmask<0x161>(v);
    v += dpp_xmask<0x162>(v);
    v += dpp_xmask<0x164>(v);
    v += dpp_xmask<0x168>(v);
    return v;
}

__global__ __launch_bounds__(NTHREAD, 1)
void fa_fwd_kernel(const float* __restrict__ Q, const float* __restrict__ K,
                   const float* __restrict__ V, float* __restrict__ O) {
    // double-buffered fp32 staging: [buf][ K(64x64) | V(64x64) ]  = 64 KB
    __shared__ __align__(16) float          sStage[2][2 * BN * Dc];
    __shared__ __align__(16) unsigned short sK [BN * Dc];          // keys bf16, row-major
    __shared__ __align__(16) unsigned short sVt[Dc * BN];          // V bf16, [d][key]
    __shared__ __align__(16) unsigned short sP [NWAVE * 16 * BN];  // per-wave P re-layout

    const int NTQ  = Nc / BM;          // 16 query tiles per head
    const int bh   = blockIdx.x / NTQ;
    const int qt   = blockIdx.x % NTQ;
    const int tid  = threadIdx.x;
    const int wave = tid >> 5;
    const int lane = tid & 31;
    const int hl   = lane >> 4;
    const int ln   = lane & 15;

    const float* Qb = Q + (size_t)bh * Nc * Dc;
    const float* Kb = K + (size_t)bh * Nc * Dc;
    const float* Vb = V + (size_t)bh * Nc * Dc;
    float*       Ob = O + (size_t)bh * Nc * Dc;

    // ---- kick off async stage of tile 0 first (overlaps with Q load) ----
    {
        float* stg0 = sStage[0];
#pragma unroll
        for (int it = 0; it < 4; ++it) {
            const int e = (tid + NTHREAD * it) * 4;
            g2l_b128(Kb + e, stg0 + e);
            g2l_b128(Vb + e, stg0 + 4096 + e);
        }
    }

    // ---- Q A-fragments for this wave's 16 rows (pre-scaled, bf16) ----
    const int    qrow = qt * BM + wave * 16 + ln;
    const float* qp   = Qb + (size_t)qrow * Dc;
    v8u qa[2];
#pragma unroll
    for (int c = 0; c < 2; ++c) {
#pragma unroll
        for (int i = 0; i < 8; ++i) {
            const int d = 32 * c + ((i < 4) ? (8 * hl + 2 * i)
                                            : (16 + 8 * hl + 2 * (i - 4)));
            const float2 qv = *(const float2*)(qp + d);
            qa[c][i] = pack2(qv.x * SCALE, qv.y * SCALE);
        }
    }

    v8f   accO[4];
    float m_i[8], l_i[8];
#pragma unroll
    for (int dj = 0; dj < 4; ++dj)
        accO[dj] = v8f{0.f, 0.f, 0.f, 0.f, 0.f, 0.f, 0.f, 0.f};
#pragma unroll
    for (int r = 0; r < 8; ++r) { m_i[r] = -1.0e30f; l_i[r] = 0.0f; }

    unsigned short* sPw = sP + wave * 16 * BN;

    for (int kt = 0; kt < NT; ++kt) {
        // ---- issue async stage of tile kt+1, then wait for tile kt ----
        if (kt + 1 < NT) {
            const float* Kt1  = Kb + (size_t)(kt + 1) * BN * Dc;
            const float* Vt1  = Vb + (size_t)(kt + 1) * BN * Dc;
            float*       stg1 = sStage[(kt + 1) & 1];
#pragma unroll
            for (int it = 0; it < 4; ++it) {
                const int e = (tid + NTHREAD * it) * 4;
                g2l_b128(Kt1 + e, stg1 + e);
                g2l_b128(Vt1 + e, stg1 + 4096 + e);
            }
            if (kt + 2 < NT) {   // L2 prefetch two tiles ahead
                __builtin_prefetch(Kt1 + BN * Dc + tid * 16, 0, 0);
                __builtin_prefetch(Vt1 + BN * Dc + tid * 16, 0, 0);
            }
            WAIT_ASYNC(8);       // in-order: tile kt's 8 copies complete
        } else {
            WAIT_ASYNC(0);
        }
        __syncthreads();         // all waves' copies visible; prev compute done

        // ---- convert pass: staged fp32 -> bf16 (V transposed) ----
        {
            const float*  stg  = sStage[kt & 1];
            unsigned int* sK32 = (unsigned int*)sK;
#pragma unroll
            for (int it = 0; it < 4; ++it) {
                const int v   = tid + NTHREAD * it;
                const int row = v >> 4;
                const int c4  = v & 15;
                const float4 kf = *(const float4*)(stg + v * 4);
                sK32[v * 2 + 0] = pack2(kf.x, kf.y);
                sK32[v * 2 + 1] = pack2(kf.z, kf.w);
                const float4 vf = *(const float4*)(stg + 4096 + v * 4);
                sVt[(c4 * 4 + 0) * BN + row] = f2bf(vf.x);
                sVt[(c4 * 4 + 1) * BN + row] = f2bf(vf.y);
                sVt[(c4 * 4 + 2) * BN + row] = f2bf(vf.z);
                sVt[(c4 * 4 + 3) * BN + row] = f2bf(vf.w);
            }
        }
        __syncthreads();

        // ---- S = Q * K^T : 4 key sub-tiles x 2 K-chunks ----
        v8f accS[4];
#pragma unroll
        for (int j = 0; j < 4; ++j) {
            v8f s = v8f{0.f, 0.f, 0.f, 0.f, 0.f, 0.f, 0.f, 0.f};
#pragma unroll
            for (int c = 0; c < 2; ++c) {
                v8u bk;
                const unsigned short* kr = sK + (j * 16 + ln) * Dc + 32 * c;
#pragma unroll
                for (int i = 0; i < 8; ++i)
                    bk[i] = *(const unsigned int*)(kr + ((i < 4) ? (8 * hl + 2 * i) : (16 + 8 * hl + 2 * (i - 4))));
                s = wmma_bf16(qa[c], bk, s);
            }
            accS[j] = s;
        }

        // ---- online softmax in C-fragment layout ----
#pragma unroll
        for (int r = 0; r < 8; ++r) {
            float mb = fmaxf(fmaxf(accS[0][r], accS[1][r]),
                             fmaxf(accS[2][r], accS[3][r]));
            mb = redmax16(mb);
            const float mn    = fmaxf(m_i[r], mb);
            const float alpha = __builtin_exp2f((m_i[r] - mn) * LOG2E);
            m_i[r] = mn;
            float rs = 0.0f;
#pragma unroll
            for (int j = 0; j < 4; ++j) {
                const float p = __builtin_exp2f((accS[j][r] - mn) * LOG2E);
                accS[j][r] = p;
                rs += p;
            }
            l_i[r] = l_i[r] * alpha + redsum16(rs);
#pragma unroll
            for (int dj = 0; dj < 4; ++dj) accO[dj][r] *= alpha;
        }

        // ---- P: C-layout -> A-layout via per-wave LDS scratch ----
#pragma unroll
        for (int j = 0; j < 4; ++j)
#pragma unroll
            for (int r = 0; r < 8; ++r)
                sPw[(r + 8 * hl) * BN + j * 16 + ln] = f2bf(accS[j][r]);
        __syncthreads();

        // ---- O += P * V ----
#pragma unroll
        for (int c = 0; c < 2; ++c) {
            v8u pa;
            const unsigned short* pr = sPw + ln * BN + 32 * c;
#pragma unroll
            for (int i = 0; i < 8; ++i) {
                const int k = (i < 4) ? (8 * hl + 2 * i)
                                      : (16 + 8 * hl + 2 * (i - 4));
                pa[i] = *(const unsigned int*)(pr + k);
            }
#pragma unroll
            for (int dj = 0; dj < 4; ++dj) {
                v8u bv;
                const unsigned short* vr = sVt + (dj * 16 + ln) * BN + 32 * c;
#pragma unroll
                for (int i = 0; i < 8; ++i)
                    bv[i] = *(const unsigned int*)(vr + ((i < 4) ? (8 * hl + 2 * i) : (16 + 8 * hl + 2 * (i - 4))));
                accO[dj] = wmma_bf16(pa, bv, accO[dj]);
            }
        }
    }

    // ---- normalize + store (fp32) ----
#pragma unroll
    for (int r = 0; r < 8; ++r) {
        const int   row = qt * BM + wave * 16 + r + 8 * hl;
        const float inv = 1.0f / l_i[r];
#pragma unroll
        for (int dj = 0; dj < 4; ++dj)
            Ob[(size_t)row * Dc + dj * 16 + ln] = accO[dj][r] * inv;
    }
}

extern "C" void kernel_launch(void* const* d_in, const int* in_sizes, int n_in,
                              void* d_out, int out_size, void* d_ws, size_t ws_size,
                              hipStream_t stream) {
    (void)in_sizes; (void)n_in; (void)out_size; (void)d_ws; (void)ws_size;
    const float* Q = (const float*)d_in[0];
    const float* K = (const float*)d_in[1];
    const float* V = (const float*)d_in[2];
    float*       O = (float*)d_out;

    const int num_blocks = Bc * Hc * (Nc / BM);   // 1024
    fa_fwd_kernel<<<num_blocks, NTHREAD, 0, stream>>>(Q, K, V, O);
}
